// GemsNetT_78477642432514
// MI455X (gfx1250) — compile-verified, matrix operand order checked by reading, not run on hardware
//
#include <hip/hip_runtime.h>
#include <hip/hip_bf16.h>
#include <math.h>

#define HID 128
#define INV2F 0.70710678118654752440f

typedef __bf16 bf16;
typedef __attribute__((ext_vector_type(16))) __bf16 v16bf;
typedef __attribute__((ext_vector_type(8)))  float  v8f;

__device__ __forceinline__ float ssilu(float x) {
  // GemNet ScaledSiLU: silu(x)/0.6
  return x * (1.6666666666666667f / (1.0f + __expf(-x)));
}

// Build one 16-lane-slice bf16 fragment per the CDNA5 16-bit 16x32 layout:
// lanes 0-15 hold K {kb+0..7, kb+16..23}, lanes 16-31 hold K {kb+8..15, kb+24..31}.
__device__ __forceinline__ v16bf frag16(const bf16* __restrict__ row, int kb, int khalf) {
  v16bf a;
  const bf16* p0 = row + kb + khalf;
  const bf16* p1 = row + kb + 16 + khalf;
#pragma unroll
  for (int j = 0; j < 8; ++j) { a[j] = p0[j]; a[j + 8] = p1[j]; }
  return a;
}

// ---------------------------------------------------------------------------
// WMMA dense: Y[M,N] = epilogue(X[M,K] @ W[K,N]).  grid=(ceil(M/64), N/16),
// block=128 (4 waves; each wave owns one 16x16 C tile, K-loop of 16x16x32 WMMAs).
// Epilogue: +bias, ScaledSiLU, *gate, (v+res)*INV2 -- compile-time selected.
// ---------------------------------------------------------------------------
template <int ACT, int BIAS, int GATE, int RES>
__global__ void k_dense_f32(const float* __restrict__ X, const float* __restrict__ W,
                            const float* __restrict__ bias, const float* __restrict__ gate,
                            const float* __restrict__ res, float* __restrict__ Y,
                            int M, int K, int N) {
  extern __shared__ bf16 smem[];
  bf16* wt = smem;            // [16][K]  (W columns of this tile, transposed)
  bf16* xt = smem + 16 * K;   // [64][K]
  const int tid = threadIdx.x;
  const int colbase = blockIdx.y * 16;
  const int r0 = blockIdx.x * 64;

  for (int i = tid; i < 16 * K; i += 128) {
    int n = i / K, k = i - n * K;
    wt[i] = (bf16)W[(size_t)k * N + colbase + n];
  }
  for (int i = tid; i < 64 * K; i += 128) {
    int r = i / K, k = i - r * K;
    int gr = r0 + r;
    xt[i] = (gr < M) ? (bf16)X[(size_t)gr * K + k] : (bf16)0.0f;
  }
  __syncthreads();

  const int lane = tid & 31;
  const int lr = (tid >> 5) * 16;
  const int n16 = lane & 15;
  const int khalf = (lane >> 4) << 3;
  const bf16* xrow = xt + (size_t)(lr + n16) * K;
  const bf16* wrow = wt + (size_t)n16 * K;

  v8f acc = {0.f, 0.f, 0.f, 0.f, 0.f, 0.f, 0.f, 0.f};
  for (int kb = 0; kb < K; kb += 32) {
    v16bf a = frag16(xrow, kb, khalf);
    v16bf b = frag16(wrow, kb, khalf);
    acc = __builtin_amdgcn_wmma_f32_16x16x32_bf16(false, a, false, b, (short)0, acc,
                                                  false, false);
  }

#pragma unroll
  for (int r = 0; r < 8; ++r) {
    int grow = r0 + lr + r + ((lane >> 4) << 3);
    if (grow >= M) continue;
    int col = colbase + n16;
    float v = acc[r];
    if (BIAS) v += bias[col];
    if (ACT) v = ssilu(v);
    size_t o = (size_t)grow * N + col;
    if (GATE) v *= gate[o];
    if (RES) v = (v + res[o]) * INV2F;
    Y[o] = v;
  }
}

// bf16-input dense (no epilogue): used for Mm(E,512) @ W2(512,64).
__global__ void k_dense_bf16(const bf16* __restrict__ X, const float* __restrict__ W,
                             float* __restrict__ Y, int M, int K, int N) {
  extern __shared__ bf16 smem[];
  bf16* wt = smem;  // [16][K]
  const int tid = threadIdx.x;
  const int colbase = blockIdx.y * 16;
  const int r0 = blockIdx.x * 64;
  for (int i = tid; i < 16 * K; i += 128) {
    int n = i / K, k = i - n * K;
    wt[i] = (bf16)W[(size_t)k * N + colbase + n];
  }
  __syncthreads();
  const int lane = tid & 31;
  const int lr = (tid >> 5) * 16;
  const int n16 = lane & 15;
  const int khalf = (lane >> 4) << 3;
  int arow = r0 + lr + n16;
  if (arow >= M) arow = M - 1;
  const bf16* xrow = X + (size_t)arow * K;
  const bf16* wrow = wt + (size_t)n16 * K;
  v8f acc = {0.f, 0.f, 0.f, 0.f, 0.f, 0.f, 0.f, 0.f};
  for (int kb = 0; kb < K; kb += 32) {
    v16bf a = frag16(xrow, kb, khalf);
    v16bf b = frag16(wrow, kb, khalf);
    acc = __builtin_amdgcn_wmma_f32_16x16x32_bf16(false, a, false, b, (short)0, acc,
                                                  false, false);
  }
#pragma unroll
  for (int r = 0; r < 8; ++r) {
    int grow = r0 + lr + r + ((lane >> 4) << 3);
    if (grow >= M) continue;
    Y[(size_t)grow * N + colbase + n16] = acc[r];
  }
}

// Small-K dense (K<=16, N=128, no act) for the RBF gate projections.
__global__ void k_dsmall(const float* __restrict__ X, const float* __restrict__ W,
                         float* __restrict__ Y, int M, int K) {
  __shared__ float w[2048];
  for (int i = threadIdx.x; i < K * 128; i += blockDim.x) w[i] = W[i];
  __syncthreads();
  int row = blockIdx.x * 2 + (threadIdx.x >> 7);
  int col = threadIdx.x & 127;
  if (row >= M) return;
  const float* x = X + (size_t)row * K;
  float s = 0.f;
  for (int k = 0; k < K; ++k) s += x[k] * w[k * 128 + col];
  Y[(size_t)row * 128 + col] = s;
}

// -------------------------- elementwise helpers ----------------------------
__global__ void k_zero(float* p, int n) {
  int i = blockIdx.x * blockDim.x + threadIdx.x;
  if (i < n) p[i] = 0.f;
}
__global__ void k_copy(const float* a, float* b, int n) {
  int i = blockIdx.x * blockDim.x + threadIdx.x;
  if (i < n) b[i] = a[i];
}
__global__ void k_mul(const float* a, const float* b, float* y, int n) {
  int i = blockIdx.x * blockDim.x + threadIdx.x;
  if (i < n) y[i] = a[i] * b[i];
}
__global__ void k_combine(const float* a, const float* b, float* y, int n) {
  int i = blockIdx.x * blockDim.x + threadIdx.x;
  if (i < n) y[i] = (a[i] + b[i]) * INV2F;
}
// xm = (xs + (x_ca + x_ac[id_swap]) * INV2) * INV2
__global__ void k_mix3(const float* xs, const float* ca, const float* ac,
                       const int* __restrict__ sw, float* y, int E) {
  int i = blockIdx.x * blockDim.x + threadIdx.x;
  if (i >= E * HID) return;
  int e = i >> 7, c = i & 127;
  float v = (ca[i] + ac[(size_t)sw[e] * HID + c]) * INV2F;
  y[i] = (xs[i] + v) * INV2F;
}
__global__ void k_vst(const float* v_ij, const float* e_ij, const float* D,
                      float* V, float* U, int E) {
  int e = blockIdx.x * blockDim.x + threadIdx.x;
  if (e >= E) return;
  float invd = 1.f / D[e];
#pragma unroll
  for (int j = 0; j < 3; ++j) {
    V[e * 3 + j] = -v_ij[e * 3 + j] * invd;
    U[e * 3 + j] = -e_ij[e * 3 + j] * invd;
  }
}
// Gaussian RBF with degree-5 polynomial envelope.
__global__ void k_rbf(const float* D, float* rbf, int E) {
  int i = blockIdx.x * blockDim.x + threadIdx.x;
  if (i >= E * HID) return;
  int e = i >> 7, r = i & 127;
  float d = D[e] * (1.0f / 6.0f);
  float d2 = d * d, d5 = d2 * d2 * d;
  float env = 1.0f - 21.0f * d5 + 35.0f * d5 * d - 15.0f * d5 * d2;
  float t = d - (float)r * (1.0f / 127.0f);
  rbf[i] = env * __expf(-8064.5f * t * t);
}
// cbf_down (7,128,16) -> Wp (128, 112) with Wp[r, s*16+c]
__global__ void k_pack_cbf(const float* src, float* Wp) {
  int i = blockIdx.x * blockDim.x + threadIdx.x;
  if (i >= 128 * 112) return;
  int r = i / 112, rem = i - r * 112;
  int s = rem >> 4, c = rem & 15;
  Wp[i] = src[s * 2048 + r * 16 + c];
}
// Legendre-based spherical basis + fused 7->16 projection (mlp_cbf_out).
__global__ void k_sph(const float* __restrict__ V, const int* __restrict__ ca,
                      const int* __restrict__ ba, const float* __restrict__ mcbf,
                      float* __restrict__ sbf, float* __restrict__ cbfo, int T) {
  __shared__ float w[112];
  if (threadIdx.x < 112) w[threadIdx.x] = mcbf[threadIdx.x];
  __syncthreads();
  int t = blockIdx.x * blockDim.x + threadIdx.x;
  if (t >= T) return;
  int a = ca[t], b = ba[t];
  float cp = V[a * 3] * V[b * 3] + V[a * 3 + 1] * V[b * 3 + 1] + V[a * 3 + 2] * V[b * 3 + 2];
  cp = fminf(1.f, fmaxf(-1.f, cp));
  float P[7];
  P[0] = 1.f; P[1] = cp;
#pragma unroll
  for (int l = 2; l < 7; ++l)
    P[l] = ((float)(2 * l - 1) * cp * P[l - 1] - (float)(l - 1) * P[l - 2]) / (float)l;
  const float C[7] = {0.28209479f, 0.48860251f, 0.63078313f, 0.74635267f,
                      0.84628438f, 0.93560258f, 1.01710724f};
  float Yl[7];
#pragma unroll
  for (int l = 0; l < 7; ++l) { Yl[l] = C[l] * P[l]; sbf[(size_t)t * 7 + l] = Yl[l]; }
#pragma unroll
  for (int c = 0; c < 16; ++c) {
    float s = 0.f;
#pragma unroll
    for (int l = 0; l < 7; ++l) s += Yl[l] * w[l * 16 + c];
    cbfo[(size_t)t * 16 + c] = s;
  }
}
__global__ void k_init_h(const int* z, const int* batch, const float* ae,
                         const float* emb, float* h, int N) {
  int i = blockIdx.x * blockDim.x + threadIdx.x;
  if (i >= N * HID) return;
  int n = i >> 7, c = i & 127;
  h[i] = ae[(size_t)z[n] * HID + c] + emb[(size_t)batch[n] * HID + c];
}
__global__ void k_concat(const float* h, const int* is, const int* it,
                         const float* third, float* cat, int E) {
  int i = blockIdx.x * blockDim.x + threadIdx.x;
  if (i >= E * 384) return;
  int e = i / 384, c = i - e * 384;
  float v;
  if (c < 128) v = h[(size_t)is[e] * HID + c];
  else if (c < 256) v = h[(size_t)it[e] * HID + c - 128];
  else v = third[(size_t)e * HID + c - 256];
  cat[i] = v;
}
__global__ void k_segsum_gate(const float* x, const float* g, const int* it,
                              float* out, int E) {
  int i = blockIdx.x * blockDim.x + threadIdx.x;
  if (i >= E * HID) return;
  int e = i >> 7, c = i & 127;
  atomicAdd(&out[(size_t)it[e] * HID + c], x[i] * g[i]);
}
// Triplet bilinear: A[s,i] = sum_k sbf[e,k,s]*xb[ba[e,k],i];
// Mm[c,i] = sum_s rbfW[e,s,c]*A[s,i]  (one wave per edge, lane = i in 0..31).
__global__ void k_trip_contract(const float* __restrict__ sbf, const float* __restrict__ rbfW,
                                const float* __restrict__ xb, const int* __restrict__ ba,
                                bf16* __restrict__ Mm, int E, int Kt) {
  int e = blockIdx.x * (blockDim.x >> 5) + (threadIdx.x >> 5);
  if (e >= E) return;
  int lane = threadIdx.x & 31;
  float A[7] = {0.f, 0.f, 0.f, 0.f, 0.f, 0.f, 0.f};
  for (int k = 0; k < Kt; ++k) {
    float g = xb[(size_t)ba[e * Kt + k] * 32 + lane];
    const float* sr = sbf + (size_t)(e * Kt + k) * 7;
#pragma unroll
    for (int s = 0; s < 7; ++s) A[s] += sr[s] * g;
  }
  const float* rw = rbfW + (size_t)e * 112;
#pragma unroll
  for (int c = 0; c < 16; ++c) {
    float acc = 0.f;
#pragma unroll
    for (int s = 0; s < 7; ++s) acc += rw[s * 16 + c] * A[s];
    Mm[(size_t)e * 512 + c * 32 + lane] = (bf16)acc;
  }
}
// Fused: trip[e,d] = sum_k (cbf_out[e,k,:] @ dcbf[:,d]) * xF[ba[e,k],d];
// out = (xF + trip) * INV2.  Avoids materializing (T,128) cbf_emb.
__global__ void k_tripF(const float* __restrict__ xF, const float* __restrict__ cbfo,
                        const float* __restrict__ dcbf, const int* __restrict__ ba,
                        float* __restrict__ out, int E, int Kt) {
  __shared__ float w[2048];
  for (int i = threadIdx.x; i < 2048; i += blockDim.x) w[i] = dcbf[i];
  __syncthreads();
  int i = blockIdx.x * blockDim.x + threadIdx.x;
  if (i >= E * HID) return;
  int e = i >> 7, d = i & 127;
  float acc = 0.f;
  for (int k = 0; k < Kt; ++k) {
    const float* cr = cbfo + (size_t)(e * Kt + k) * 16;
    float s = 0.f;
#pragma unroll
    for (int c = 0; c < 16; ++c) s += cr[c] * w[c * 128 + d];
    acc += s * xF[(size_t)ba[e * Kt + k] * HID + d];
  }
  out[i] = (xF[i] + acc) * INV2F;
}
// dst[row] += X[row,:] @ w  (128 -> 1 projection; accumulated across out-blocks)
__global__ void k_rowdot(const float* __restrict__ X, const float* __restrict__ w,
                         float* __restrict__ dst, int M) {
  int row = blockIdx.x * (blockDim.x >> 5) + (threadIdx.x >> 5);
  if (row >= M) return;
  int lane = threadIdx.x & 31;
  float s = 0.f;
#pragma unroll
  for (int j = 0; j < 4; ++j) {
    int c = lane + 32 * j;
    s += X[(size_t)row * HID + c] * w[c];
  }
  for (int off = 16; off; off >>= 1) s += __shfl_xor(s, off, 32);
  if (lane == 0) dst[row] += s;
}
__global__ void k_Eg(const float* Et, const int* batch, float* Eg, int N) {
  int n = blockIdx.x * blockDim.x + threadIdx.x;
  if (n < N) atomicAdd(&Eg[batch[n]], Et[n]);
}
__global__ void k_divEg(float* Eg, const int* na, int B) {
  int b = blockIdx.x * blockDim.x + threadIdx.x;
  if (b < B) Eg[b] /= (float)na[b];
}
__global__ void k_force(const float* Fst, const float* U, const int* it, float* Ft, int E) {
  int e = blockIdx.x * blockDim.x + threadIdx.x;
  if (e >= E) return;
  float f = Fst[e];
#pragma unroll
  for (int j = 0; j < 3; ++j) atomicAdd(&Ft[(size_t)it[e] * 3 + j], f * U[e * 3 + j]);
}
__global__ void k_pos(const float* x, const float* Ft, float* pos, int n3) {
  int i = blockIdx.x * blockDim.x + threadIdx.x;
  if (i >= n3) return;
  float p = x[i] + Ft[i];
  pos[i] = p - floorf(p);
}

// ---------------------------------------------------------------------------
extern "C" void kernel_launch(void* const* d_in, const int* in_sizes, int n_in,
                              void* d_out, int out_size, void* d_ws, size_t ws_size,
                              hipStream_t stream) {
  (void)n_in; (void)out_size; (void)ws_size;
  const int N = in_sizes[0];
  const int E = in_sizes[2];
  const int T = in_sizes[5];
  const int B = in_sizes[7];
  const int Kt = T / E;

  const int* z      = (const int*)d_in[0];
  const int* batch  = (const int*)d_in[1];
  const int* idx_s  = (const int*)d_in[2];
  const int* idx_t  = (const int*)d_in[3];
  const int* idswap = (const int*)d_in[4];
  const int* id3_ba = (const int*)d_in[5];
  const int* id3_ca = (const int*)d_in[6];
  const int* natoms = (const int*)d_in[7];
  const float* D    = (const float*)d_in[8];
  const float* v_ij = (const float*)d_in[9];
  const float* e_ij = (const float*)d_in[10];
  const float* xpos = (const float*)d_in[11];
  const float* emb  = (const float*)d_in[12];

  // params: jax pytree order (sorted dict keys; lists/tuples in order)
  auto PF = [&](int i) { return (const float*)d_in[i]; };
  const float* atom_emb   = PF(13);
  const float* cbf_down   = PF(14);
  const float* edge_emb_b = PF(15);
  const float* edge_emb_w = PF(16);
  struct IntBlk {
    const float *after[4], *au_in, *au_rbf, *au_res[6], *before[2];
    const float *concat_b, *concat_w, *dense_ca, *res_m[2];
    const float *W2, *dense_ba, *down, *mlp_rbf, *up_ac, *up_ca;
  } IB[3];
  int base = 17;
  for (int i = 0; i < 3; ++i) {
    for (int j = 0; j < 4; ++j) IB[i].after[j] = PF(base + j);
    IB[i].au_in = PF(base + 4); IB[i].au_rbf = PF(base + 5);
    for (int j = 0; j < 6; ++j) IB[i].au_res[j] = PF(base + 6 + j);
    IB[i].before[0] = PF(base + 12); IB[i].before[1] = PF(base + 13);
    IB[i].concat_b = PF(base + 14); IB[i].concat_w = PF(base + 15);
    IB[i].dense_ca = PF(base + 16);
    IB[i].res_m[0] = PF(base + 17); IB[i].res_m[1] = PF(base + 18);
    IB[i].W2 = PF(base + 19); IB[i].dense_ba = PF(base + 20);
    IB[i].down = PF(base + 21); IB[i].mlp_rbf = PF(base + 22);
    IB[i].up_ac = PF(base + 23); IB[i].up_ca = PF(base + 24);
    base += 25;
  }
  const float* mlp_cbf_out = PF(base + 0);
  const float* mlp_rbf3    = PF(base + 1);
  const float* mlp_rbf_h   = PF(base + 2);
  const float* mlp_rbf_out = PF(base + 3);
  base += 4;
  struct OutBlk {
    const float *dense_cbf_F, *dense_rbf, *dense_rbf_F, *out_E, *out_F;
    const float *seqE_in, *seqE_res[6], *seqF_in, *seqF_res[6];
  } OB[4];
  for (int i = 0; i < 4; ++i) {
    OB[i].dense_cbf_F = PF(base + 0); OB[i].dense_rbf = PF(base + 1);
    OB[i].dense_rbf_F = PF(base + 2); OB[i].out_E = PF(base + 3);
    OB[i].out_F = PF(base + 4); OB[i].seqE_in = PF(base + 5);
    for (int j = 0; j < 6; ++j) OB[i].seqE_res[j] = PF(base + 6 + j);
    OB[i].seqF_in = PF(base + 12);
    for (int j = 0; j < 6; ++j) OB[i].seqF_res[j] = PF(base + 13 + j);
    base += 19;
  }

  // ---- workspace carve -----------------------------------------------------
  char* wp = (char*)d_ws;
  size_t off = 0;
  auto alloc = [&](size_t bytes) {
    void* p = wp + off;
    off += (bytes + 255) & ~(size_t)255;
    return p;
  };
  float* V      = (float*)alloc((size_t)E * 3 * 4);
  float* U      = (float*)alloc((size_t)E * 3 * 4);
  float* rbf    = (float*)alloc((size_t)E * HID * 4);
  float* rbf3   = (float*)alloc((size_t)E * 16 * 4);
  float* rbfh16 = (float*)alloc((size_t)E * 16 * 4);
  float* rbfo16 = (float*)alloc((size_t)E * 16 * 4);
  float* rbfW   = (float*)alloc((size_t)E * 112 * 4);
  float* sbf    = (float*)alloc((size_t)T * 7 * 4);
  float* cbfo   = (float*)alloc((size_t)T * 16 * 4);
  float* h      = (float*)alloc((size_t)N * HID * 4);
  float* m      = (float*)alloc((size_t)E * HID * 4);
  float* mB     = (float*)alloc((size_t)E * HID * 4);
  float* t0     = (float*)alloc((size_t)E * HID * 4);
  float* t1     = (float*)alloc((size_t)E * HID * 4);
  float* t2     = (float*)alloc((size_t)E * HID * 4);
  float* t3     = (float*)alloc((size_t)E * HID * 4);
  float* g1b    = (float*)alloc((size_t)E * HID * 4);
  char*  catR   = (char*)alloc((size_t)E * 1536);  // cat | {xb, Mm(bf16), t64}
  float* cat    = (float*)catR;
  float* xb     = (float*)catR;
  bf16*  Mm     = (bf16*)(catR + (size_t)E * 128);
  float* t64    = (float*)(catR + (size_t)E * 1152);
  float* aN     = (float*)alloc((size_t)N * HID * 4);
  float* hnA    = (float*)alloc((size_t)N * HID * 4);
  float* hnB    = (float*)alloc((size_t)N * HID * 4);
  float* Et     = (float*)alloc((size_t)N * 4);
  float* Fst    = (float*)alloc((size_t)E * 4);
  float* Wcbf   = (float*)alloc((size_t)128 * 112 * 4);

  float* out    = (float*)d_out;
  float* outEg  = out + (size_t)N * HID;
  float* outPos = outEg + B;
  float* outFt  = outPos + (size_t)N * 3;

  auto blk1 = [](int n) { return dim3((unsigned)((n + 255) / 256)); };
  auto zero = [&](float* p, int n) { k_zero<<<blk1(n), 256, 0, stream>>>(p, n); };

  enum { M_NOACT = 0, M_ACT, M_ACT_BIAS, M_ACT_GATE, M_ACT_RES };
  auto dense = [&](const float* X, const float* W, const float* bias,
                   const float* gate, const float* res, float* Y,
                   int M, int K, int Ncols, int mode) {
    dim3 g((unsigned)((M + 63) / 64), (unsigned)(Ncols / 16));
    dim3 b(128);
    size_t sh = (size_t)80 * K * sizeof(bf16);
    switch (mode) {
      case M_NOACT:
        k_dense_f32<0, 0, 0, 0><<<g, b, sh, stream>>>(X, W, bias, gate, res, Y, M, K, Ncols);
        break;
      case M_ACT:
        k_dense_f32<1, 0, 0, 0><<<g, b, sh, stream>>>(X, W, bias, gate, res, Y, M, K, Ncols);
        break;
      case M_ACT_BIAS:
        k_dense_f32<1, 1, 0, 0><<<g, b, sh, stream>>>(X, W, bias, gate, res, Y, M, K, Ncols);
        break;
      case M_ACT_GATE:
        k_dense_f32<1, 0, 1, 0><<<g, b, sh, stream>>>(X, W, bias, gate, res, Y, M, K, Ncols);
        break;
      case M_ACT_RES:
        k_dense_f32<1, 0, 0, 1><<<g, b, sh, stream>>>(X, W, bias, gate, res, Y, M, K, Ncols);
        break;
    }
  };
  // residual block: Y = (X + act(act(X@W0)@W1)) * INV2
  auto resid = [&](const float* Xin, float* tmp, float* Yout,
                   const float* W0, const float* W1, int M) {
    dense(Xin, W0, nullptr, nullptr, nullptr, tmp, M, HID, HID, M_ACT);
    dense(tmp, W1, nullptr, nullptr, Xin, Yout, M, HID, HID, M_ACT_RES);
  };
  auto dsmall = [&](const float* X, const float* W, float* Y, int M, int K) {
    k_dsmall<<<dim3((unsigned)((M + 1) / 2)), 256, 0, stream>>>(X, W, Y, M, K);
  };

  // ---- setup ---------------------------------------------------------------
  zero(Et, N); zero(Fst, E); zero(outEg, B); zero(outFt, N * 3);
  k_vst<<<blk1(E), 256, 0, stream>>>(v_ij, e_ij, D, V, U, E);
  k_rbf<<<blk1(E * HID), 256, 0, stream>>>(D, rbf, E);
  k_pack_cbf<<<blk1(128 * 112), 256, 0, stream>>>(cbf_down, Wcbf);
  dense(rbf, mlp_rbf3,    nullptr, nullptr, nullptr, rbf3,   E, HID, 16,  M_NOACT);
  dense(rbf, mlp_rbf_h,   nullptr, nullptr, nullptr, rbfh16, E, HID, 16,  M_NOACT);
  dense(rbf, mlp_rbf_out, nullptr, nullptr, nullptr, rbfo16, E, HID, 16,  M_NOACT);
  dense(rbf, Wcbf,        nullptr, nullptr, nullptr, rbfW,   E, HID, 112, M_NOACT);
  k_sph<<<blk1(T), 256, 0, stream>>>(V, id3_ca, id3_ba, mlp_cbf_out, sbf, cbfo, T);
  k_init_h<<<blk1(N * HID), 256, 0, stream>>>(z, batch, atom_emb, emb, h, N);
  k_concat<<<blk1(E * 384), 256, 0, stream>>>(h, idx_s, idx_t, rbf, cat, E);
  dense(cat, edge_emb_w, edge_emb_b, nullptr, nullptr, m, E, 384, HID, M_ACT_BIAS);

  auto outBlock = [&](const OutBlk& p) {
    // energy branch
    dsmall(rbfo16, p.dense_rbf, g1b, E, 16);
    zero(aN, N * HID);
    k_segsum_gate<<<blk1(E * HID), 256, 0, stream>>>(m, g1b, idx_t, aN, E);
    dense(aN, p.seqE_in, nullptr, nullptr, nullptr, hnA, N, HID, HID, M_ACT);
    resid(hnA, hnB, aN,  p.seqE_res[0], p.seqE_res[1], N);
    resid(aN,  hnB, hnA, p.seqE_res[2], p.seqE_res[3], N);
    resid(hnA, hnB, aN,  p.seqE_res[4], p.seqE_res[5], N);
    k_rowdot<<<dim3((unsigned)((N + 3) / 4)), 128, 0, stream>>>(aN, p.out_E, Et, N);
    // force branch
    dense(m, p.seqF_in, nullptr, nullptr, nullptr, t0, E, HID, HID, M_ACT);
    resid(t0, t1, t2, p.seqF_res[0], p.seqF_res[1], E);
    resid(t2, t1, t0, p.seqF_res[2], p.seqF_res[3], E);
    resid(t0, t1, t2, p.seqF_res[4], p.seqF_res[5], E);
    dsmall(rbfo16, p.dense_rbf_F, g1b, E, 16);
    k_mul<<<blk1(E * HID), 256, 0, stream>>>(t2, g1b, t2, E * HID);
    k_tripF<<<blk1(E * HID), 256, 0, stream>>>(t2, cbfo, p.dense_cbf_F, id3_ba, t0, E, Kt);
    k_rowdot<<<dim3((unsigned)((E + 3) / 4)), 128, 0, stream>>>(t0, p.out_F, Fst, E);
  };

  auto intBlock = [&](const IntBlk& p) {
    dense(m, p.dense_ca, nullptr, nullptr, nullptr, t0, E, HID, HID, M_ACT);
    // triplet interaction
    dsmall(rbf3, p.mlp_rbf, g1b, E, 16);
    dense(m, p.dense_ba, nullptr, g1b, nullptr, t1, E, HID, HID, M_ACT_GATE);
    dense(t1, p.down, nullptr, nullptr, nullptr, xb, E, HID, 32, M_ACT);
    k_trip_contract<<<dim3((unsigned)((E + 7) / 8)), 256, 0, stream>>>(
        sbf, rbfW, xb, id3_ba, Mm, E, Kt);
    k_dense_bf16<<<dim3((unsigned)((E + 63) / 64), 4), 128,
                   (size_t)16 * 512 * sizeof(bf16), stream>>>(Mm, p.W2, t64, E, 512, 64);
    dense(t64, p.up_ca, nullptr, nullptr, nullptr, t1, E, 64, HID, M_ACT);
    dense(t64, p.up_ac, nullptr, nullptr, nullptr, t2, E, 64, HID, M_ACT);
    k_mix3<<<blk1(E * HID), 256, 0, stream>>>(t0, t1, t2, idswap, t3, E);
    // before / skip / after
    resid(t3, t1, t0, p.before[0], p.before[1], E);
    k_combine<<<blk1(E * HID), 256, 0, stream>>>(t0, m, t3, E * HID);
    resid(t3, t1, t0, p.after[0], p.after[1], E);
    resid(t0, t1, t3, p.after[2], p.after[3], E);   // xm final in t3
    // atom update
    dsmall(rbfh16, p.au_rbf, g1b, E, 16);
    zero(aN, N * HID);
    k_segsum_gate<<<blk1(E * HID), 256, 0, stream>>>(t3, g1b, idx_t, aN, E);
    dense(aN, p.au_in, nullptr, nullptr, nullptr, hnA, N, HID, HID, M_ACT);
    resid(hnA, hnB, aN,  p.au_res[0], p.au_res[1], N);
    resid(aN,  hnB, hnA, p.au_res[2], p.au_res[3], N);
    resid(hnA, hnB, aN,  p.au_res[4], p.au_res[5], N);
    k_combine<<<blk1(N * HID), 256, 0, stream>>>(h, aN, h, N * HID);
    // edge update
    k_concat<<<blk1(E * 384), 256, 0, stream>>>(h, idx_s, idx_t, t3, cat, E);
    dense(cat, p.concat_w, p.concat_b, nullptr, nullptr, mB, E, 384, HID, M_ACT_BIAS);
    resid(mB, t1, t0, p.res_m[0], p.res_m[1], E);   // new m in t0
    float* nm = t0; t0 = mB; mB = m; m = nm;
  };

  outBlock(OB[0]);
  for (int i = 0; i < 3; ++i) {
    intBlock(IB[i]);
    outBlock(OB[i + 1]);
  }

  // ---- finalize ------------------------------------------------------------
  k_Eg<<<blk1(N), 256, 0, stream>>>(Et, batch, outEg, N);
  k_divEg<<<blk1(B), 256, 0, stream>>>(outEg, natoms, B);
  k_force<<<blk1(E), 256, 0, stream>>>(Fst, U, idx_t, outFt, E);
  k_pos<<<blk1(N * 3), 256, 0, stream>>>(xpos, outFt, outPos, N * 3);
  k_copy<<<blk1(N * HID), 256, 0, stream>>>(h, out, N * HID);
}
